// Qwen3VLVisualAdapter_77988016161346
// MI455X (gfx1250) — matrix operand order repaired and checked by reading, not compiled, not run on hardware
//
#include <hip/hip_runtime.h>
#include <hip/hip_bf16.h>
#include <math.h>

// Qwen3VL visual adapter, fused: LayerNorm -> down(3584->224) -> exact GELU
// -> up(224->3584) -> x + alpha*res.  fp32 in/out, f16 WMMA with fp32 accum.
// Weights pre-converted to f16 in d_ws by a prep kernel each launch.
// Weight staging uses CDNA5 async global->LDS copies (ASYNCcnt-tracked).

#define HIDDEN  3584
#define BOT     224
#define NTILES  (BOT / 16)     // 14 N-tiles in the bottleneck dim
#define MTILE   64             // tokens per block
#define KCHUNK  64             // K columns staged per down-GEMM chunk
#define NWAVES  4              // waves per block (wave32)
#define BLOCK   (NWAVES * 32)  // 128 threads
#define GTILE   4              // up-phase N-tiles (of 16 hidden cols) per LDS stage
#define KSTEPS  (BOT / 32)     // 7 K-steps of 32 in the up GEMM
#define WTOTAL  (BOT * HIDDEN) // 802816 elements per weight matrix

#if defined(__has_builtin)
#if __has_builtin(__builtin_amdgcn_global_load_async_to_lds_b128)
#define USE_ASYNC_LDS 1
#endif
#endif
#ifndef USE_ASYNC_LDS
#define USE_ASYNC_LDS 0
#endif

typedef __attribute__((ext_vector_type(16))) _Float16 v16h;
typedef __attribute__((ext_vector_type(8)))  _Float16 v8h;
typedef __attribute__((ext_vector_type(8)))  float    v8f;
typedef __attribute__((ext_vector_type(4)))  float    v4f;
typedef __attribute__((ext_vector_type(4)))  unsigned int v4u;
typedef int v4i __attribute__((vector_size(16)));

union FragH { v16h v; v8h h[2]; _Float16 e[16]; };
union Pack8 { _Float16 e[8]; v4u q; };
union Pack4 { _Float16 e[4]; unsigned long long q; };

__device__ __forceinline__ float gelu_exact(float v) {
    return 0.5f * v * (1.0f + erff(v * 0.70710678118654752f));
}

#if USE_ASYNC_LDS
// Builtin signature (from probe diagnostic): param0 = AS1 v4i* (global src),
// param1 = AS3 v4i* (LDS dst), then imm offset, imm cpol.
__device__ __forceinline__ void async_copy_b128(const _Float16* gsrc, _Float16* ldst) {
    __builtin_amdgcn_global_load_async_to_lds_b128(
        (__attribute__((address_space(1))) v4i*)(gsrc),
        (__attribute__((address_space(3))) v4i*)(ldst),
        0, 0);
}
__device__ __forceinline__ void async_wait0() {
#if __has_builtin(__builtin_amdgcn_s_wait_asynccnt)
    __builtin_amdgcn_s_wait_asynccnt(0);
#else
    asm volatile("s_wait_asynccnt 0x0" ::: "memory");
#endif
}
#endif

// ---------------- Prep: fp32 -> f16 weight conversion into d_ws -----------
// Combined space: [0, WTOTAL) = w_down, [WTOTAL, 2*WTOTAL) = w_up.
__global__ __launch_bounds__(256)
void convert_weights_kernel(const float* __restrict__ wdown,
                            const float* __restrict__ wup,
                            _Float16* __restrict__ wdown_h,
                            _Float16* __restrict__ wup_h)
{
    const int base = (blockIdx.x * 256 + threadIdx.x) * 4;   // 4 elems/thread
    const float*  src = (base < WTOTAL) ? (wdown + base) : (wup + (base - WTOTAL));
    _Float16*     dst = (base < WTOTAL) ? (wdown_h + base) : (wup_h + (base - WTOTAL));
    v4f v = *(const v4f*)src;
    Pack4 p;
    #pragma unroll
    for (int k = 0; k < 4; ++k) p.e[k] = (_Float16)v[k];
    *(unsigned long long*)dst = p.q;
}

// ---------------- Main fused kernel ---------------------------------------
__global__ __launch_bounds__(BLOCK)
void adapter_fused_kernel(const float* __restrict__ x,
                          const float* __restrict__ gamma,
                          const float* __restrict__ beta,
                          const _Float16* __restrict__ wdown_h, // [BOT][HIDDEN] f16
                          const _Float16* __restrict__ wup_h,   // [HIDDEN][BOT] f16
                          const float* __restrict__ alphap,
                          float* __restrict__ out)
{
    // LDS: union of down-phase (A 64x64 + B 224x64 = 18432 halfs) and
    //      up-phase   (H 64x224 + W 64x224       = 28672 halfs) regions.
    __shared__ _Float16 smem[MTILE * BOT + GTILE * 16 * BOT]; // 28672 halfs = 56 KB
    __shared__ float sMean[MTILE];
    __shared__ float sRstd[MTILE];

    _Float16* As = smem;                   // [MTILE][KCHUNK]
    _Float16* Bs = smem + MTILE * KCHUNK;  // [BOT][KCHUNK]
    _Float16* Hs = smem;                   // [MTILE][BOT]
    _Float16* Wb = smem + MTILE * BOT;     // [GTILE*16][BOT]

    const int tid  = threadIdx.x;
    const int wave = tid >> 5;
    const int lane = tid & 31;
    const int l15  = lane & 15;
    const int tok0 = blockIdx.x * MTILE;
    const float alpha = *alphap;

    // ================= Phase 0: per-token mean / rstd (float4 loads) ======
    for (int t = 0; t < 16; ++t) {
        const int tl = wave * 16 + t;
        const float* row = x + (size_t)(tok0 + tl) * HIDDEN;
        float s = 0.f, s2 = 0.f;
        for (int i = lane * 4; i < HIDDEN; i += 128) {
            v4f v = *(const v4f*)(row + i);
            #pragma unroll
            for (int k = 0; k < 4; ++k) { s += v[k]; s2 += v[k] * v[k]; }
        }
        for (int o = 16; o > 0; o >>= 1) {
            s  += __shfl_down(s,  o, 32);
            s2 += __shfl_down(s2, o, 32);
        }
        if (lane == 0) {
            float mu  = s  * (1.0f / HIDDEN);
            float var = s2 * (1.0f / HIDDEN) - mu * mu;
            sMean[tl] = mu;
            sRstd[tl] = rsqrtf(var + 1e-5f);
        }
    }
    __syncthreads();

    // ================= Phase 1: down GEMM (16x224 per wave) ===============
    v8f acc[NTILES];
    #pragma unroll
    for (int j = 0; j < NTILES; ++j) acc[j] = (v8f){0.f,0.f,0.f,0.f,0.f,0.f,0.f,0.f};

    for (int kbase = 0; kbase < HIDDEN; kbase += KCHUNK) {
        // Stage w_down slice (already f16): async global->LDS
        #pragma unroll 2
        for (int idx = tid; idx < (BOT * KCHUNK) / 8; idx += BLOCK) {
            const int r  = idx >> 3;
            const int c8 = (idx & 7) * 8;
            const _Float16* src = wdown_h + (size_t)r * HIDDEN + kbase + c8;
            _Float16*       dst = Bs + r * KCHUNK + c8;
#if USE_ASYNC_LDS
            async_copy_b128(src, dst);
#else
            *(v4u*)dst = *(const v4u*)src;
#endif
        }
        // Stage normalized activations (f16): 8 cols / thread, packed stores
        for (int idx = tid; idx < (MTILE * KCHUNK) / 8; idx += BLOCK) {
            const int t  = idx >> 3;
            const int c8 = (idx & 7) * 8;
            const int col = kbase + c8;
            const float mu = sMean[t], rs = sRstd[t];
            v4f xv0 = *(const v4f*)(x + (size_t)(tok0 + t) * HIDDEN + col);
            v4f xv1 = *(const v4f*)(x + (size_t)(tok0 + t) * HIDDEN + col + 4);
            v4f g0  = *(const v4f*)(gamma + col);
            v4f g1  = *(const v4f*)(gamma + col + 4);
            v4f b0  = *(const v4f*)(beta + col);
            v4f b1  = *(const v4f*)(beta + col + 4);
            Pack8 p;
            #pragma unroll
            for (int k = 0; k < 4; ++k) {
                p.e[k]     = (_Float16)((xv0[k] - mu) * rs * g0[k] + b0[k]);
                p.e[4 + k] = (_Float16)((xv1[k] - mu) * rs * g1[k] + b1[k]);
            }
            *(v4u*)(As + t * KCHUNK + c8) = p.q;
        }
        // Hint: prefetch next K-chunk of w_down (rows 0..127)
        if (kbase + KCHUNK < HIDDEN) {
            __builtin_prefetch(wdown_h + (size_t)tid * HIDDEN + kbase + KCHUNK, 0, 0);
        }
#if USE_ASYNC_LDS
        async_wait0();
#endif
        __syncthreads();

        #pragma unroll
        for (int kk = 0; kk < KCHUNK / 32; ++kk) {
            // A fragment: 16x32 f16 (half-waves K 0-15 vs 16-31, matching B)
            const _Float16* arow = &As[(wave * 16 + l15) * KCHUNK + kk * 32];
            const int aoff = (lane < 16) ? 0 : 16;
            FragH a;
            a.h[0] = *(const v8h*)(arow + aoff);
            a.h[1] = *(const v8h*)(arow + aoff + 8);

            const _Float16* bbase = &Bs[l15 * KCHUNK + kk * 32 + ((lane < 16) ? 0 : 16)];
            FragH bb[2];
            bb[0].h[0] = *(const v8h*)(bbase);
            bb[0].h[1] = *(const v8h*)(bbase + 8);
            #pragma unroll
            for (int j = 0; j < NTILES; ++j) {
                if (j + 1 < NTILES) {
                    const _Float16* bp = bbase + (size_t)(j + 1) * 16 * KCHUNK;
                    bb[(j + 1) & 1].h[0] = *(const v8h*)(bp);
                    bb[(j + 1) & 1].h[1] = *(const v8h*)(bp + 8);
                }
                acc[j] = __builtin_amdgcn_wmma_f32_16x16x32_f16(
                    false, a.v, false, bb[j & 1].v, (short)0, acc[j], false, false);
            }
        }
        __syncthreads();
    }

    // ================= Phase 2: GELU + spill H (f16) to LDS ===============
    {
        const int mb = (lane < 16) ? 0 : 8;
        #pragma unroll
        for (int j = 0; j < NTILES; ++j) {
            const int n = j * 16 + l15;
            #pragma unroll
            for (int i = 0; i < 8; ++i) {
                float v = gelu_exact(acc[j][i]);
                Hs[(wave * 16 + mb + i) * BOT + n] = (_Float16)v;
            }
        }
    }
    __syncthreads();

    // Build the 7 up-GEMM A-fragments (16x32 slices of H) once per wave
    v16h ha[KSTEPS];
    {
        const _Float16* hrow = &Hs[(wave * 16 + l15) * BOT];
        const int aoff = (lane < 16) ? 0 : 16;
        #pragma unroll
        for (int ks = 0; ks < KSTEPS; ++ks) {
            FragH a;
            a.h[0] = *(const v8h*)(hrow + ks * 32 + aoff);
            a.h[1] = *(const v8h*)(hrow + ks * 32 + aoff + 8);
            ha[ks] = a.v;
        }
    }

    // ================= Phase 3: up GEMM + residual ========================
    for (int hbase = 0; hbase < HIDDEN; hbase += GTILE * 16) {
        // Stage GTILE*16 rows of w_up (f16): async global->LDS
        #pragma unroll 2
        for (int idx = tid; idx < (GTILE * 16 * BOT) / 8; idx += BLOCK) {
            const int hl = idx / (BOT / 8);
            const int c8 = (idx % (BOT / 8)) * 8;
            const _Float16* src = wup_h + (size_t)(hbase + hl) * BOT + c8;
            _Float16*       dst = Wb + hl * BOT + c8;
#if USE_ASYNC_LDS
            async_copy_b128(src, dst);
#else
            *(v4u*)dst = *(const v4u*)src;
#endif
        }
#if USE_ASYNC_LDS
        async_wait0();
#endif
        __syncthreads();

        #pragma unroll
        for (int jj = 0; jj < GTILE; ++jj) {
            v8f uacc = (v8f){0.f,0.f,0.f,0.f,0.f,0.f,0.f,0.f};
            const _Float16* wrow = &Wb[(jj * 16 + l15) * BOT + ((lane < 16) ? 0 : 16)];
            FragH bb[2];
            bb[0].h[0] = *(const v8h*)(wrow);
            bb[0].h[1] = *(const v8h*)(wrow + 8);
            #pragma unroll
            for (int ks = 0; ks < KSTEPS; ++ks) {
                if (ks + 1 < KSTEPS) {
                    const _Float16* bp = wrow + (ks + 1) * 32;
                    bb[(ks + 1) & 1].h[0] = *(const v8h*)(bp);
                    bb[(ks + 1) & 1].h[1] = *(const v8h*)(bp + 8);
                }
                uacc = __builtin_amdgcn_wmma_f32_16x16x32_f16(
                    false, ha[ks], false, bb[ks & 1].v, (short)0, uacc, false, false);
            }
            // residual + store (fp32)
            const int h  = hbase + jj * 16 + l15;
            const int mb = (lane < 16) ? 0 : 8;
            #pragma unroll
            for (int i = 0; i < 8; ++i) {
                const size_t o = (size_t)(tok0 + wave * 16 + mb + i) * HIDDEN + h;
                out[o] = x[o] + alpha * uacc[i];
            }
        }
        __syncthreads();
    }
}

extern "C" void kernel_launch(void* const* d_in, const int* in_sizes, int n_in,
                              void* d_out, int out_size, void* d_ws, size_t ws_size,
                              hipStream_t stream) {
    const float* x      = (const float*)d_in[0];
    const float* gamma  = (const float*)d_in[1];
    const float* beta   = (const float*)d_in[2];
    const float* wdown  = (const float*)d_in[3];
    const float* wup    = (const float*)d_in[4];
    const float* alphap = (const float*)d_in[5];
    float* out = (float*)d_out;

    // f16 weight images in workspace (needs 2 * WTOTAL * 2 = ~3.2 MB)
    _Float16* wdown_h = (_Float16*)d_ws;
    _Float16* wup_h   = wdown_h + (size_t)WTOTAL;

    const int conv_grid = (2 * WTOTAL) / (256 * 4);   // 1568 blocks, exact
    convert_weights_kernel<<<conv_grid, 256, 0, stream>>>(wdown, wup, wdown_h, wup_h);

    const int tokens = 4 * 4096;              // B * S
    const int grid   = tokens / MTILE;        // 256 blocks
    adapter_fused_kernel<<<grid, BLOCK, 0, stream>>>(
        x, gamma, beta, wdown_h, wup_h, alphap, out);
}
